// CycleEmbedding0_30382598652489
// MI455X (gfx1250) — matrix operand order repaired and checked by reading, not compiled, not run on hardware
//
#include <hip/hip_runtime.h>

typedef __attribute__((ext_vector_type(16))) _Float16 v16h;
typedef __attribute__((ext_vector_type(8)))  _Float16 v8h;
typedef __attribute__((ext_vector_type(8)))  float    v8f;
typedef __attribute__((ext_vector_type(4)))  unsigned v4u;

#define HIDDEN 128
#define KPAD   32

// ---------------- Pass 0: zero the counts workspace (uint4 stores) ----------
__global__ __launch_bounds__(256) void ce_zero_kernel(unsigned* __restrict__ p, size_t n4) {
    size_t i = (size_t)blockIdx.x * blockDim.x + threadIdx.x;
    if (i < n4) {
        v4u z = {0u, 0u, 0u, 0u};
        ((v4u*)p)[i] = z;
    }
}

// ---------------- Pass 1: build hi/lo f16 split of emb_W, K-transposed ------
// embHiT/embLoT layout: [HIDDEN][KPAD] f16, so a lane's 16 K-values are contiguous.
__global__ __launch_bounds__(256) void ce_prep_emb_kernel(const float* __restrict__ embW,
                                                          _Float16* __restrict__ embHiT,
                                                          _Float16* __restrict__ embLoT,
                                                          int vocab) {
    int idx = blockIdx.x * blockDim.x + threadIdx.x;   // 16 blocks * 256 = 4096 = 128*32
    int n = idx >> 5;          // hidden index 0..127
    int k = idx & 31;          // padded vocab index 0..31
    float v = (k < vocab) ? embW[k * HIDDEN + n] : 0.0f;
    _Float16 hi = (_Float16)v;
    _Float16 lo = (_Float16)(v - (float)hi);
    embHiT[idx] = hi;
    embLoT[idx] = lo;
}

// ---------------- Pass 2: histogram H[c, v] += 1 (u32 atomics, no return) ---
__global__ __launch_bounds__(256) void ce_count_kernel(const int* __restrict__ x,
                                                       const int* __restrict__ atom_to_cycle,
                                                       unsigned* __restrict__ counts,
                                                       int n_pairs) {
    int p = blockIdx.x * blockDim.x + threadIdx.x;
    if (p < n_pairs) {
        int a = atom_to_cycle[p];            // row 0: atom indices
        int c = atom_to_cycle[n_pairs + p];  // row 1: cycle indices
        int v = x[a];
        atomicAdd(counts + (size_t)c * KPAD + v, 1u);
    }
}

// ---------------- Pass 3: out = H @ (emb_hi + emb_lo) via v_wmma ------------
// Block = one 16-row tile of cycles; wave w = column tile w (HIDDEN/16 = 8 waves).
__global__ __launch_bounds__(256) void ce_gemm_kernel(const unsigned* __restrict__ counts,
                                                      const _Float16* __restrict__ embHiT,
                                                      const _Float16* __restrict__ embLoT,
                                                      float* __restrict__ out,
                                                      int n_cycles) {
    const int lane = threadIdx.x & 31;
    const int wave = threadIdx.x >> 5;      // 0..7 -> column tile
    const int rowTile = blockIdx.x;
    const int r16  = lane & 15;
    const int half = lane >> 4;             // 0: lanes 0-15, 1: lanes 16-31

    // ---- A matrix (16x32 f16): M = r16, K = kA + j, kA = 16*half
    const unsigned* crow = counts + (size_t)(rowTile * 16 + r16) * KPAD;
    const int kA = half * 16;
    v4u c0 = *(const v4u*)(crow + kA);          // K = kA .. kA+3
    v4u c1 = *(const v4u*)(crow + kA + 4);      // K = kA+4 .. kA+7
    v4u c2 = *(const v4u*)(crow + kA + 8);      // K = kA+8 .. kA+11
    v4u c3 = *(const v4u*)(crow + kA + 12);     // K = kA+12 .. kA+15
    v16h A;
#pragma unroll
    for (int i = 0; i < 4; ++i) {
        A[i]      = (_Float16)(float)c0[i];
        A[4 + i]  = (_Float16)(float)c1[i];
        A[8 + i]  = (_Float16)(float)c2[i];
        A[12 + i] = (_Float16)(float)c3[i];
    }

    // ---- B matrices (32x16 f16): N = r16, K = kB + j, kB = 16*half
    const int col = wave * 16 + r16;
    const int kB  = half * 16;
    const _Float16* bh = embHiT + (size_t)col * KPAD + kB;
    const _Float16* bl = embLoT + (size_t)col * KPAD + kB;
    v8h bh0 = *(const v8h*)(bh);
    v8h bh1 = *(const v8h*)(bh + 8);
    v8h bl0 = *(const v8h*)(bl);
    v8h bl1 = *(const v8h*)(bl + 8);
    v16h Bh, Bl;
#pragma unroll
    for (int j = 0; j < 8; ++j) {
        Bh[j] = bh0[j]; Bh[8 + j] = bh1[j];
        Bl[j] = bl0[j]; Bl[8 + j] = bl1[j];
    }

    // ---- D = A*Bh + A*Bl + 0, f32 accumulate
    v8f acc = {};
    acc = __builtin_amdgcn_wmma_f32_16x16x32_f16(false, A, false, Bh,
                                                 (short)0, acc, false, false);
    acc = __builtin_amdgcn_wmma_f32_16x16x32_f16(false, A, false, Bl,
                                                 (short)0, acc, false, false);

    // ---- C/D layout: N = r16, VGPR r -> M = r + 8*half
    // Single 64-bit base + constant byte offsets (r*HIDDEN*4 = r*512 fits the
    // 24-bit immediate). Tile-partial check is uniform per block: only the
    // last row tile can be partial, so the fast path is 8 branch-free stores.
    const int mBase = rowTile * 16 + half * 8;
    float* po = out + (size_t)mBase * HIDDEN + col;
    if (rowTile * 16 + 16 <= n_cycles) {
#pragma unroll
        for (int r = 0; r < 8; ++r)
            po[(size_t)r * HIDDEN] = acc[r];
    } else {
#pragma unroll
        for (int r = 0; r < 8; ++r)
            if (mBase + r < n_cycles)
                po[(size_t)r * HIDDEN] = acc[r];
    }
}

extern "C" void kernel_launch(void* const* d_in, const int* in_sizes, int n_in,
                              void* d_out, int out_size, void* d_ws, size_t ws_size,
                              hipStream_t stream) {
    const int*   x         = (const int*)d_in[0];
    const int*   a2c       = (const int*)d_in[1];
    const float* embW      = (const float*)d_in[2];
    const int    n_pairs   = in_sizes[1] / 2;
    const int    vocab     = in_sizes[2] / HIDDEN;
    const int    n_cycles  = out_size / HIDDEN;
    const int    rowTiles  = (n_cycles + 15) / 16;

    // Workspace layout: [counts: rowTiles*16*KPAD u32][embHiT: 4096 f16][embLoT: 4096 f16]
    unsigned* counts = (unsigned*)d_ws;
    size_t countsElems = (size_t)rowTiles * 16 * KPAD;
    size_t off = (countsElems * sizeof(unsigned) + 255) & ~(size_t)255;
    _Float16* embHiT = (_Float16*)((char*)d_ws + off);
    _Float16* embLoT = embHiT + HIDDEN * KPAD;

    // Pass 0: zero counts (must happen every call — graph replays don't re-poison)
    size_t n4 = countsElems / 4;
    ce_zero_kernel<<<(unsigned)((n4 + 255) / 256), 256, 0, stream>>>(counts, n4);

    // Pass 1: hi/lo f16 split of emb_W (transposed for contiguous K loads)
    ce_prep_emb_kernel<<<(HIDDEN * KPAD) / 256, 256, 0, stream>>>(embW, embHiT, embLoT, vocab);

    // Pass 2: histogram over pairs
    ce_count_kernel<<<(n_pairs + 255) / 256, 256, 0, stream>>>(x, a2c, counts, n_pairs);

    // Pass 3: WMMA GEMM -> out
    ce_gemm_kernel<<<rowTiles, 256, 0, stream>>>(counts, embHiT, embLoT, (float*)d_out, n_cycles);
}
